// MyModel_39505109188981
// MI455X (gfx1250) — compile-verified, matrix operand order checked by reading, not run on hardware
//
#include <hip/hip_runtime.h>
#include <hip/hip_bf16.h>

typedef __attribute__((ext_vector_type(16))) _Float16 v16h;
typedef __attribute__((ext_vector_type(8)))  _Float16 v8h;
typedef __attribute__((ext_vector_type(8)))  float    v8f;

#define N_NODES  200000
#define N_EDGES  3200000
#define N_GRAPHS 2000
#define DIM      128
#define IN_DIM   77
#define BN_EPS   1e-5f

#define LDS_K    136   // 128 + 8 halfs pad: keeps 16B alignment, kills bank conflicts

// ---------------------------------------------------------------------------
// zero fill
__global__ void zero_kernel(float* __restrict__ p, long long n) {
    long long i = (long long)blockIdx.x * blockDim.x + threadIdx.x;
    if (i < n) p[i] = 0.0f;
}

// ---------------------------------------------------------------------------
// edge scatter-add: agg[dst] += in[src]  (one wave of 32 lanes per edge,
// each lane covers 4 feature dims -> global_atomic_add_f32, L2 resident)
__global__ void scatter_add_kernel(const int* __restrict__ src,
                                   const int* __restrict__ dst,
                                   const float* __restrict__ in, int lda,
                                   int Kdim, float* __restrict__ agg) {
    long long tid = (long long)blockIdx.x * blockDim.x + threadIdx.x;
    int e    = (int)(tid >> 5);
    int lane = (int)(tid & 31);
    if (e >= N_EDGES) return;
    int k = lane * 4;
    if (k >= Kdim) return;
    int s = src[e], d = dst[e];
    const float* ip = in  + (size_t)s * lda + k;
    float*       op = agg + ((size_t)d << 7) + k;   // agg stride fixed 128
    int rem = Kdim - k; if (rem > 4) rem = 4;
    #pragma unroll 4
    for (int j = 0; j < rem; ++j) atomicAdd(op + j, ip[j]);
}

// ---------------------------------------------------------------------------
// Build a 16-half WMMA fragment from two aligned 16B LDS runs.
// Per ISA 16-bit A/B layout: lane half i -> K = kb + (i<8 ? i : i+8),
// i.e. two contiguous runs of 8 halfs at +0 and +16.
__device__ __forceinline__ v16h frag_load(const _Float16* p) {
    v8h lo = *(const v8h*)p;          // K offsets kb+0..7
    v8h hi = *(const v8h*)(p + 16);   // K offsets kb+16..23
    return __builtin_shufflevector(lo, hi,
        0, 1, 2, 3, 4, 5, 6, 7, 8, 9, 10, 11, 12, 13, 14, 15);
}

// ---------------------------------------------------------------------------
// WMMA MLP: out = relu((A [+ Aadd]) @ W + bias), W is [Kdim x 128] f32.
// Block = 256 threads (8 waves) covering 64 rows x 128 cols.
// Weights staged TRANSPOSED ([n][k]) so B fragments are two b128 LDS loads;
// each wave loads its B fragments once and reuses them across 4 M tiles.
__global__ void __launch_bounds__(256)
wmma_mlp_kernel(const float* __restrict__ A,    int lda,
                const float* __restrict__ Aadd, int ldadd,
                const float* __restrict__ W,    int Kdim,
                const float* __restrict__ bias,
                float* __restrict__ out, int ldo) {
    __shared__ _Float16 sWt[128 * LDS_K];  // [n][k]  ~34 KB
    __shared__ _Float16 sA [64  * LDS_K];  // [m][k]  ~17 KB
    const int tid = threadIdx.x;
    const int m0  = blockIdx.x * 64;

    // stage weights transposed (coalesced global reads over n), zero-pad K
    for (int idx = tid; idx < 128 * 128; idx += 256) {
        int k = idx >> 7, n = idx & 127;
        float v = (k < Kdim) ? W[k * 128 + n] : 0.0f;
        sWt[n * LDS_K + k] = (_Float16)v;
    }
    // stage A tile (64 x 128), fused optional add (GIN self term), zero-pad K
    for (int idx = tid; idx < 64 * 128; idx += 256) {
        int r = idx >> 7, k = idx & 127;
        float v = 0.0f;
        if (k < Kdim) {
            size_t row = (size_t)(m0 + r);
            v = A[row * lda + k];
            if (Aadd) v += Aadd[row * ldadd + k];
        }
        sA[r * LDS_K + k] = (_Float16)v;
    }
    __syncthreads();

    const int wave    = tid >> 5;
    const int lane    = tid & 31;
    const int half_hi = lane >> 4;        // 0: lanes 0-15, 1: lanes 16-31
    const int l15     = lane & 15;
    const int kb      = half_hi * 8;      // K sub-base per 16-bit A/B layout
    const int cn      = wave * 16;        // this wave's N tile

    // B fragments: loaded once, reused for all 4 M tiles
    v16h bf[4];
    #pragma unroll
    for (int kc = 0; kc < 4; ++kc)
        bf[kc] = frag_load(&sWt[(cn + l15) * LDS_K + kc * 32 + kb]);

    #pragma unroll
    for (int mt = 0; mt < 4; ++mt) {
        const _Float16* parow = &sA[(mt * 16 + l15) * LDS_K + kb];
        v8f acc = {};
        #pragma unroll
        for (int kc = 0; kc < 4; ++kc) {
            v16h af = frag_load(parow + kc * 32);
            acc = __builtin_amdgcn_wmma_f32_16x16x32_f16(
                      false, af, false, bf[kc], (short)0, acc, false, false);
        }
        // C/D layout: VGPR r holds M = r + 8*half_hi, N = l15
        #pragma unroll
        for (int r = 0; r < 8; ++r) {
            int row = m0 + mt * 16 + r + half_hi * 8;
            int col = cn + l15;
            float v = acc[r] + bias[col];
            out[(size_t)row * ldo + col] = fmaxf(v, 0.0f);
        }
    }
}

// ---------------------------------------------------------------------------
// BN stats: per-block (100 rows) channel sum / sumsq, atomic into stats[256]
__global__ void bn_stats_kernel(const float* __restrict__ u, int ldu,
                                float* __restrict__ stats) {
    int c  = threadIdx.x;            // 128 channels
    int r0 = blockIdx.x * 100;
    float s = 0.0f, q = 0.0f;
    for (int r = 0; r < 100; ++r) {
        float v = u[(size_t)(r0 + r) * ldu + c];
        s += v; q += v * v;
    }
    atomicAdd(&stats[c], s);
    atomicAdd(&stats[128 + c], q);
}

__global__ void bn_finalize_kernel(const float* __restrict__ stats,
                                   float* __restrict__ murs) {
    int c = threadIdx.x;             // 128
    const float invN = 1.0f / (float)N_NODES;
    float mu  = stats[c] * invN;
    float var = stats[128 + c] * invN - mu * mu;
    murs[c]       = mu;
    murs[128 + c] = rsqrtf(var + BN_EPS);
}

__global__ void bn_apply_kernel(float* __restrict__ u, int ldu,
                                const float* __restrict__ murs,
                                const float* __restrict__ gamma,
                                const float* __restrict__ beta) {
    int idx = blockIdx.x * 256 + threadIdx.x;   // exactly N_NODES*128
    int c = idx & 127, r = idx >> 7;
    float* p = u + (size_t)r * ldu + c;
    float v = *p;
    *p = (v - murs[c]) * murs[128 + c] * gamma[c] + beta[c];
}

// ---------------------------------------------------------------------------
// segment max pool: exactly 100 contiguous nodes per graph
__global__ void pool_max_kernel(const float* __restrict__ nrep,
                                float* __restrict__ pooled) {
    int g = blockIdx.x;
    int c = blockIdx.y * 128 + threadIdx.x;     // 3 * 128 = 384
    const float* base = nrep + (size_t)g * 100 * 384 + c;
    float m = -3.402823e38f;
    for (int r = 0; r < 100; ++r) m = fmaxf(m, base[(size_t)r * 384]);
    pooled[(size_t)g * 384 + c] = m;
}

// ---------------------------------------------------------------------------
// fused head: relu(pooled@ew+eb) -> elu(@w1+b1) -> elu(@w2+b2) -> @w3+b3
__global__ void __launch_bounds__(256)
head_kernel(const float* __restrict__ pooled,
            const float* __restrict__ ew, const float* __restrict__ eb,
            const float* __restrict__ w1, const float* __restrict__ b1,
            const float* __restrict__ w2, const float* __restrict__ b2,
            const float* __restrict__ w3, const float* __restrict__ b3,
            float* __restrict__ out) {
    __shared__ float s_in[384], s_e[256], s_a[512], s_b[512], red[256];
    int g = blockIdx.x, t = threadIdx.x;
    for (int i = t; i < 384; i += 256) s_in[i] = pooled[(size_t)g * 384 + i];
    __syncthreads();
    {   // e = relu(in @ ew + eb), ew [384,256]
        float s = eb[t];
        for (int k = 0; k < 384; ++k) s += s_in[k] * ew[k * 256 + t];
        s_e[t] = fmaxf(s, 0.0f);
    }
    __syncthreads();
    for (int c = t; c < 512; c += 256) {   // elu(e @ w1 + b1), w1 [256,512]
        float s = b1[c];
        for (int k = 0; k < 256; ++k) s += s_e[k] * w1[k * 512 + c];
        s_a[c] = (s > 0.0f) ? s : (expf(s) - 1.0f);
    }
    __syncthreads();
    for (int c = t; c < 512; c += 256) {   // elu(a @ w2 + b2), w2 [512,512]
        float s = b2[c];
        for (int k = 0; k < 512; ++k) s += s_a[k] * w2[k * 512 + c];
        s_b[c] = (s > 0.0f) ? s : (expf(s) - 1.0f);
    }
    __syncthreads();
    float p = 0.0f;                        // b @ w3 + b3, w3 [512,1]
    for (int k = t; k < 512; k += 256) p += s_b[k] * w3[k];
    red[t] = p; __syncthreads();
    for (int off = 128; off > 0; off >>= 1) {
        if (t < off) red[t] += red[t + off];
        __syncthreads();
    }
    if (t == 0) out[g] = red[0] + b3[0];
}

// ---------------------------------------------------------------------------
extern "C" void kernel_launch(void* const* d_in, const int* in_sizes, int n_in,
                              void* d_out, int out_size, void* d_ws, size_t ws_size,
                              hipStream_t stream) {
    (void)in_sizes; (void)n_in; (void)out_size; (void)ws_size;

    const float* x   = (const float*)d_in[0];
    const int*   ei  = (const int*)d_in[1];
    const int*   src = ei;
    const int*   dst = ei + N_EDGES;
    // layer params: base = 3 + 6*L : w1, b1, w2, b2, gamma, beta
    const float *W1[3], *B1[3], *W2[3], *B2[3], *G[3], *Bt[3];
    for (int L = 0; L < 3; ++L) {
        int b = 3 + 6 * L;
        W1[L] = (const float*)d_in[b + 0];
        B1[L] = (const float*)d_in[b + 1];
        W2[L] = (const float*)d_in[b + 2];
        B2[L] = (const float*)d_in[b + 3];
        G [L] = (const float*)d_in[b + 4];
        Bt[L] = (const float*)d_in[b + 5];
    }
    const float* emb_w = (const float*)d_in[21];
    const float* emb_b = (const float*)d_in[22];
    const float* rw1   = (const float*)d_in[23];
    const float* rb1   = (const float*)d_in[24];
    const float* rw2   = (const float*)d_in[25];
    const float* rb2   = (const float*)d_in[26];
    const float* rw3   = (const float*)d_in[27];
    const float* rb3   = (const float*)d_in[28];
    float* out = (float*)d_out;

    // workspace layout (floats)
    float* ws     = (float*)d_ws;
    float* agg    = ws;                                   // 200000*128
    float* t      = agg  + (size_t)N_NODES * 128;         // 200000*128
    float* nrep   = t    + (size_t)N_NODES * 128;         // 200000*384
    float* pooled = nrep + (size_t)N_NODES * 384;         // 2000*384
    float* stats  = pooled + (size_t)N_GRAPHS * 384;      // 256
    float* murs   = stats + 256;                          // 256

    const long long aggN   = (long long)N_NODES * 128;
    const long long edgeT  = (long long)N_EDGES * 32;
    const int aggBlocks    = (int)((aggN + 255) / 256);
    const int edgeBlocks   = (int)((edgeT + 255) / 256);
    const int mlpBlocks    = N_NODES / 64;                // 3125, exact
    const int bnBlocks     = (int)(aggN / 256);           // 100000, exact

    for (int L = 0; L < 3; ++L) {
        const float* in; int lda, Kdim;
        if (L == 0) { in = x;                    lda = IN_DIM; Kdim = IN_DIM; }
        else        { in = nrep + (L - 1) * 128; lda = 384;    Kdim = 128;    }

        zero_kernel<<<aggBlocks, 256, 0, stream>>>(agg, aggN);
        scatter_add_kernel<<<edgeBlocks, 256, 0, stream>>>(src, dst, in, lda, Kdim, agg);
        // t = relu((agg + in) @ W1 + b1)
        wmma_mlp_kernel<<<mlpBlocks, 256, 0, stream>>>(agg, 128, in, lda,
                                                       W1[L], Kdim, B1[L], t, 128);
        // node_rep slice = relu(t @ W2 + b2)  (pre-BN)
        wmma_mlp_kernel<<<mlpBlocks, 256, 0, stream>>>(t, 128, nullptr, 0,
                                                       W2[L], 128, B2[L],
                                                       nrep + L * 128, 384);
        zero_kernel<<<1, 256, 0, stream>>>(stats, 256);
        bn_stats_kernel<<<N_GRAPHS, 128, 0, stream>>>(nrep + L * 128, 384, stats);
        bn_finalize_kernel<<<1, 128, 0, stream>>>(stats, murs);
        bn_apply_kernel<<<bnBlocks, 256, 0, stream>>>(nrep + L * 128, 384,
                                                      murs, G[L], Bt[L]);
    }

    pool_max_kernel<<<dim3(N_GRAPHS, 3), 128, 0, stream>>>(nrep, pooled);
    head_kernel<<<N_GRAPHS, 256, 0, stream>>>(pooled, emb_w, emb_b,
                                              rw1, rb1, rw2, rb2, rw3, rb3, out);
}
